// NodeLevel_88295937671213
// MI455X (gfx1250) — compile-verified, matrix operand order checked by reading, not run on hardware
//
#include <hip/hip_runtime.h>

typedef __attribute__((ext_vector_type(16))) _Float16 v16h;
typedef __attribute__((ext_vector_type(8)))  float    v8f;

#define NN   10000
#define EE   320000
#define DIN_ 256
#define BN_EPS 1e-5f
#define INV_TAU 20.0f   // 1/0.05

static inline int cdiv(int a, int b) { return (a + b - 1) / b; }

// ---------------- device helpers ----------------

// Load one 16x32 f16 fragment (A or B operand) from a row-major [rows][ld] f16
// matrix. Per ISA 7.12.2: lane half 0 holds K in [0,8) U [16,24), half 1 holds
// K in [8,16) U [24,32); row/col = lane & 15. Two 128-bit loads per lane.
__device__ inline v16h load_frag_f16(const _Float16* __restrict__ base, int ld,
                                     int row0, int k0) {
  const int lane = threadIdx.x & 31;
  const int h = lane >> 4;
  const _Float16* p = base + (size_t)(row0 + (lane & 15)) * ld + k0 + 8 * h;
  union { v16h v; float4 f[2]; } u;
  u.f[0] = *(const float4*)p;         // K: k0+8h .. +8
  u.f[1] = *(const float4*)(p + 16);  // K: k0+16+8h .. +8
  return u.v;
}

__device__ inline v8f wmma32(v16h a, v16h b, v8f c) {
  return __builtin_amdgcn_wmma_f32_16x16x32_f16(
      /*neg_a=*/false, a, /*neg_b=*/false, b,
      /*c_mod=*/(short)0, c, /*reuse_a=*/false, /*reuse_b=*/false);
}

// ---------------- elementwise / prep kernels ----------------

__global__ void f2h_kernel(const float* __restrict__ x, _Float16* __restrict__ y, int n) {
  int t = blockIdx.x * 256 + threadIdx.x;
  if (t < n) y[t] = (_Float16)x[t];
}

// Wt[nc][k] = (f16) W[k][nc]
__global__ void wtrans_kernel(const float* __restrict__ W, _Float16* __restrict__ Wt,
                              int k, int nc) {
  int t = blockIdx.x * 256 + threadIdx.x;
  if (t >= k * nc) return;
  int kk = t / nc, n = t % nc;
  Wt[(size_t)n * k + kk] = (_Float16)W[t];
}

__global__ void set_kernel(float* __restrict__ p, float v, int n) {
  int t = blockIdx.x * 256 + threadIdx.x;
  if (t < n) p[t] = v;
}

__global__ void deg_edge_kernel(const int* __restrict__ dst, float* __restrict__ deg, int e) {
  int t = blockIdx.x * 256 + threadIdx.x;
  if (t < e) atomicAdd(&deg[dst[t]], 1.0f);
}

__global__ void rsqrt_kernel(float* __restrict__ p, int n) {
  int t = blockIdx.x * 256 + threadIdx.x;
  if (t < n) p[t] = rsqrtf(p[t]);
}

// ---------------- WMMA GEMM: C[M][Nc] = A[M][K] * Bt[Nc][K]^T (+bias) ----------------
// block = 256 threads = 8 waves (2x4), wave tile 32x32, block tile 64x128.
__global__ __launch_bounds__(256)
void wmma_gemm_kernel(const _Float16* __restrict__ A, const _Float16* __restrict__ Bt,
                      const float* __restrict__ bias, float* __restrict__ C,
                      int M, int K, int Nc) {
  const int lane = threadIdx.x & 31;
  const int wave = threadIdx.x >> 5;
  const int wm = wave >> 2, wn = wave & 3;
  const int r0 = blockIdx.x * 64 + wm * 32;
  const int c0 = blockIdx.y * 128 + wn * 32;
  int rb[2] = { r0, r0 + 16 };
  int cb[2] = { c0, c0 + 16 };
  int rl[2] = { rb[0] < M ? rb[0] : 0, rb[1] < M ? rb[1] : 0 };   // clamp for loads
  int cl[2] = { cb[0] < Nc ? cb[0] : 0, cb[1] < Nc ? cb[1] : 0 };
  v8f acc[2][2] = {};
  for (int k = 0; k < K; k += 32) {
    v16h a0 = load_frag_f16(A, K, rl[0], k);
    v16h a1 = load_frag_f16(A, K, rl[1], k);
    v16h b0 = load_frag_f16(Bt, K, cl[0], k);
    v16h b1 = load_frag_f16(Bt, K, cl[1], k);
    acc[0][0] = wmma32(a0, b0, acc[0][0]);
    acc[0][1] = wmma32(a0, b1, acc[0][1]);
    acc[1][0] = wmma32(a1, b0, acc[1][0]);
    acc[1][1] = wmma32(a1, b1, acc[1][1]);
  }
  const int colf = lane & 15;
  const int rowh = (lane >> 4) * 8;
#pragma unroll
  for (int mi = 0; mi < 2; ++mi) {
    if (rb[mi] >= M) continue;
#pragma unroll
    for (int ni = 0; ni < 2; ++ni) {
      if (cb[ni] >= Nc) continue;
      int col = cb[ni] + colf;
      float bv = bias ? bias[col] : 0.0f;
#pragma unroll
      for (int v = 0; v < 8; ++v) {
        int row = rb[mi] + rowh + v;
        C[(size_t)row * Nc + col] = acc[mi][ni][v] + bv;
      }
    }
  }
}

// ---------------- GCN aggregation ----------------

__global__ void agg_self_kernel(const float* __restrict__ xw, const float* __restrict__ dinv,
                                const float* __restrict__ bias, float* __restrict__ out,
                                int n, int dout) {
  int t = blockIdx.x * 256 + threadIdx.x;
  if (t >= n * dout) return;
  int i = t / dout, c = t % dout;
  float di = dinv[i];
  out[t] = xw[t] * di * di + bias[c];
}

__global__ void agg_edge_kernel(const float* __restrict__ xw, const float* __restrict__ dinv,
                                const int* __restrict__ src, const int* __restrict__ dst,
                                float* __restrict__ out, int e, int dout) {
  long long t = (long long)blockIdx.x * 256 + threadIdx.x;
  if (t >= (long long)e * dout) return;
  int ei = (int)(t / dout), c = (int)(t % dout);
  int s = src[ei], d = dst[ei];
  atomicAdd(&out[(size_t)d * dout + c], xw[(size_t)s * dout + c] * dinv[s] * dinv[d]);
}

// ---------------- BatchNorm (training stats) + PReLU ----------------
// One block per 32 columns; 8 row-groups of 32 lanes each, coalesced reads.
__global__ __launch_bounds__(256)
void bn_stats_kernel(const float* __restrict__ x, float* __restrict__ mean,
                     float* __restrict__ var, int n, int nc) {
  const int lane = threadIdx.x & 31;
  const int rg = threadIdx.x >> 5;
  const int c = blockIdx.x * 32 + lane;
  float s = 0.f, ss = 0.f;
  for (int r = rg; r < n; r += 8) {
    float v = x[(size_t)r * nc + c];
    s += v; ss += v * v;
  }
  __shared__ float sh[2][8][32];
  sh[0][rg][lane] = s; sh[1][rg][lane] = ss;
  __syncthreads();
  if (rg == 0) {
    for (int k = 1; k < 8; ++k) { s += sh[0][k][lane]; ss += sh[1][k][lane]; }
    float mu = s / n;
    mean[c] = mu;
    var[c] = ss / n - mu * mu;
  }
}

__global__ void bn_prelu_kernel(float* __restrict__ x, const float* __restrict__ mean,
                                const float* __restrict__ var, const float* __restrict__ gamma,
                                const float* __restrict__ beta, const float* __restrict__ aP,
                                int n, int nc) {
  int t = blockIdx.x * 256 + threadIdx.x;
  if (t >= n * nc) return;
  int c = t % nc;
  float y = (x[t] - mean[c]) * rsqrtf(var[c] + BN_EPS) * gamma[c] + beta[c];
  x[t] = y >= 0.f ? y : aP[0] * y;
}

// ---------------- L2 normalize rows -> f16 ----------------
__global__ __launch_bounds__(256)
void l2norm_half_kernel(const float* __restrict__ h, _Float16* __restrict__ z, int n, int d) {
  const int lane = threadIdx.x & 31;
  const int wave = threadIdx.x >> 5;
  int row = blockIdx.x * 8 + wave;
  if (row >= n) return;
  const float* p = h + (size_t)row * d;
  float ss = 0.f;
  for (int c = lane; c < d; c += 32) { float v = p[c]; ss += v * v; }
  ss += __shfl_xor(ss, 1); ss += __shfl_xor(ss, 2); ss += __shfl_xor(ss, 4);
  ss += __shfl_xor(ss, 8); ss += __shfl_xor(ss, 16);
  float nrm = sqrtf(ss);
  nrm = nrm < 1e-12f ? 1e-12f : nrm;
  float inv = 1.0f / nrm;
  for (int c = lane; c < d; c += 32) z[(size_t)row * d + c] = (_Float16)(p[c] * inv);
}

// ---------------- Fused similarity: exp((ZA ZB^T)/tau) row/col sums + diag ----------------
// Same tiling as wmma_gemm_kernel; K fixed to 128. N x N matrix never touches memory.
__global__ __launch_bounds__(256)
void sim_exp_kernel(const _Float16* __restrict__ ZA, const _Float16* __restrict__ ZB,
                    float* __restrict__ rowsum, float* __restrict__ colsum,
                    float* __restrict__ diag, int n) {
  const int lane = threadIdx.x & 31;
  const int wave = threadIdx.x >> 5;
  const int wm = wave >> 2, wn = wave & 3;
  const int r0 = blockIdx.x * 64 + wm * 32;
  const int c0 = blockIdx.y * 128 + wn * 32;
  int rb[2] = { r0, r0 + 16 };
  int cb[2] = { c0, c0 + 16 };
  int rl[2] = { rb[0] < n ? rb[0] : 0, rb[1] < n ? rb[1] : 0 };
  int cl[2] = { cb[0] < n ? cb[0] : 0, cb[1] < n ? cb[1] : 0 };
  v8f acc[2][2] = {};
#pragma unroll
  for (int k = 0; k < 128; k += 32) {
    v16h a0 = load_frag_f16(ZA, 128, rl[0], k);
    v16h a1 = load_frag_f16(ZA, 128, rl[1], k);
    v16h b0 = load_frag_f16(ZB, 128, cl[0], k);
    v16h b1 = load_frag_f16(ZB, 128, cl[1], k);
    acc[0][0] = wmma32(a0, b0, acc[0][0]);
    acc[0][1] = wmma32(a0, b1, acc[0][1]);
    acc[1][0] = wmma32(a1, b0, acc[1][0]);
    acc[1][1] = wmma32(a1, b1, acc[1][1]);
  }
  const int halfId = lane >> 4;
#pragma unroll
  for (int mi = 0; mi < 2; ++mi) {
    if (rb[mi] >= n) continue;
#pragma unroll
    for (int ni = 0; ni < 2; ++ni) {
      if (cb[ni] >= n) continue;
      float cs = 0.f;
#pragma unroll
      for (int v = 0; v < 8; ++v) {
        // C layout: lane -> col = lane&15, row = v + 8*(lane>>4)
        float e = __expf(acc[mi][ni][v] * INV_TAU);
        cs += e;
        float rsum = e;                             // reduce across the 16 lanes of this half
        rsum += __shfl_xor(rsum, 1);
        rsum += __shfl_xor(rsum, 2);
        rsum += __shfl_xor(rsum, 4);
        rsum += __shfl_xor(rsum, 8);
        if ((lane & 15) == v)
          atomicAdd(&rowsum[rb[mi] + v + 8 * halfId], rsum);
        if (rb[mi] == cb[ni]) {                     // diagonal fragment
          if (lane == v)      diag[rb[mi] + v]     = e;   // m in 0..7
          if (lane == v + 24) diag[rb[mi] + 8 + v] = e;   // m in 8..15
        }
      }
      cs += __shfl_xor(cs, 16);                     // combine the two 8-row halves per column
      if (colsum && lane < 16) atomicAdd(&colsum[cb[ni] + lane], cs);
    }
  }
}

// ---------------- Final loss reduction ----------------
__global__ __launch_bounds__(256)
void loss_kernel(const float* __restrict__ rs1, const float* __restrict__ dg1,
                 const float* __restrict__ rs2, const float* __restrict__ dg2,
                 const float* __restrict__ rsb, const float* __restrict__ csb,
                 const float* __restrict__ dgb, float* __restrict__ out, int n) {
  __shared__ float sh[256];
  float acc = 0.f;
  for (int i = threadIdx.x; i < n; i += 256) {
    float db = dgb[i];
    float l1 = -__logf(db / (rs1[i] + rsb[i] - dg1[i]));
    float l2 = -__logf(db / (rs2[i] + csb[i] - dg2[i]));
    acc += 0.5f * (l1 + l2);
  }
  sh[threadIdx.x] = acc;
  __syncthreads();
  for (int s = 128; s > 0; s >>= 1) {
    if (threadIdx.x < s) sh[threadIdx.x] += sh[threadIdx.x + s];
    __syncthreads();
  }
  if (threadIdx.x == 0) out[0] = sh[0] / n;
}

// ---------------- host side ----------------

struct EncL { const float *W, *b, *gamma, *beta, *a; };
struct ProjP { const float *W1, *b1, *gamma, *beta, *a, *W2, *b2; };

static void mapEncLayer(void* const* din, const int* sz, int base, EncL& L) {
  auto F = [&](int i) { return (const float*)din[base + i]; };
  if (sz[base + 1] == 1) { L.W = F(0); L.a = F(1); L.b = F(2); L.beta = F(3); L.gamma = F(4); }
  else                   { L.W = F(0); L.b = F(1); L.gamma = F(2); L.beta = F(3); L.a = F(4); }
}
static void mapProj(void* const* din, const int* sz, int base, ProjP& P) {
  auto F = [&](int i) { return (const float*)din[base + i]; };
  if (sz[base + 1] > 10000) { P.W1 = F(0); P.W2 = F(1); P.a = F(2); P.b1 = F(3); P.b2 = F(4); P.beta = F(5); P.gamma = F(6); }
  else                      { P.W1 = F(0); P.b1 = F(1); P.gamma = F(2); P.beta = F(3); P.a = F(4); P.W2 = F(5); P.b2 = F(6); }
}

extern "C" void kernel_launch(void* const* d_in, const int* in_sizes, int n_in,
                              void* d_out, int out_size, void* d_ws, size_t ws_size,
                              hipStream_t stream) {
  // ---- map inputs (handle insertion-order, jax-sorted, and concatenated layouts) ----
  const float *x1, *x2, *x;
  const int *ei1, *ei2, *ei;
  EncL sEnc[2], tEnc[2];
  ProjP sProj, tProj;

  if (n_in == 10) {  // each top-level value concatenated flat (insertion order inside)
    x1 = (const float*)d_in[0]; x2 = (const float*)d_in[1]; x = (const float*)d_in[2];
    ei1 = (const int*)d_in[3]; ei2 = (const int*)d_in[4]; ei = (const int*)d_in[5];
    auto encFlat = [&](const float* base, EncL L[2]) {
      const float* p = base;
      L[0].W = p; p += 256 * 256; L[0].b = p; p += 256; L[0].gamma = p; p += 256;
      L[0].beta = p; p += 256; L[0].a = p; p += 1;
      L[1].W = p; p += 256 * 128; L[1].b = p; p += 128; L[1].gamma = p; p += 128;
      L[1].beta = p; p += 128; L[1].a = p;
    };
    auto projFlat = [&](const float* base, ProjP& P) {
      const float* p = base;
      P.W1 = p; p += 128 * 512; P.b1 = p; p += 512; P.gamma = p; p += 512;
      P.beta = p; p += 512; P.a = p; p += 1; P.W2 = p; p += 512 * 128; P.b2 = p;
    };
    encFlat((const float*)d_in[6], sEnc);
    encFlat((const float*)d_in[7], tEnc);
    projFlat((const float*)d_in[8], sProj);
    projFlat((const float*)d_in[9], tProj);
  } else if (in_sizes[0] == 2 * EE) {  // alphabetical top-level
    ei = (const int*)d_in[0]; ei1 = (const int*)d_in[1]; ei2 = (const int*)d_in[2];
    mapEncLayer(d_in, in_sizes, 3, sEnc[0]);  mapEncLayer(d_in, in_sizes, 8, sEnc[1]);
    mapProj(d_in, in_sizes, 13, sProj);
    mapEncLayer(d_in, in_sizes, 20, tEnc[0]); mapEncLayer(d_in, in_sizes, 25, tEnc[1]);
    mapProj(d_in, in_sizes, 30, tProj);
    x = (const float*)d_in[37]; x1 = (const float*)d_in[38]; x2 = (const float*)d_in[39];
  } else {  // insertion order (as written in setup_inputs)
    x1 = (const float*)d_in[0]; x2 = (const float*)d_in[1]; x = (const float*)d_in[2];
    ei1 = (const int*)d_in[3]; ei2 = (const int*)d_in[4]; ei = (const int*)d_in[5];
    mapEncLayer(d_in, in_sizes, 6, sEnc[0]);  mapEncLayer(d_in, in_sizes, 11, sEnc[1]);
    mapEncLayer(d_in, in_sizes, 16, tEnc[0]); mapEncLayer(d_in, in_sizes, 21, tEnc[1]);
    mapProj(d_in, in_sizes, 26, sProj);
    mapProj(d_in, in_sizes, 33, tProj);
  }

  // ---- workspace bump allocator ----
  char* wsp = (char*)d_ws;
  auto alloc = [&](size_t bytes) -> void* {
    void* p = wsp;
    wsp += (bytes + 255) & ~(size_t)255;
    return p;
  };
  _Float16* hA   = (_Float16*)alloc((size_t)NN * 512 * 2);   // f16 staging for GEMM A
  float*    xw   = (float*)alloc((size_t)NN * 256 * 4);      // X*W buffer
  float*    agg  = (float*)alloc((size_t)NN * 256 * 4);      // aggregation / layer buffer
  float*    hid  = xw;                                       // projector hidden (N x 512), aliases xw+agg
  float*    hencS = (float*)alloc((size_t)NN * 128 * 4);
  float*    hencT = (float*)alloc((size_t)NN * 128 * 4);
  float*    h1   = (float*)alloc((size_t)NN * 128 * 4);
  float*    h2   = (float*)alloc((size_t)NN * 128 * 4);
  float*    dinv = (float*)alloc((size_t)NN * 4);
  float*    mean = (float*)alloc(512 * 4);
  float*    var  = (float*)alloc(512 * 4);
  float*    lvec = (float*)alloc((size_t)7 * NN * 4);
  float *rs1 = lvec, *dg1 = lvec + NN, *rs2 = lvec + 2 * NN, *dg2 = lvec + 3 * NN,
        *rsb = lvec + 4 * NN, *csb = lvec + 5 * NN, *dgb = lvec + 6 * NN;
  _Float16* hz1 = (_Float16*)alloc((size_t)NN * 128 * 2);
  _Float16* hz2 = (_Float16*)alloc((size_t)NN * 128 * 2);
  _Float16* sW0t = (_Float16*)alloc((size_t)256 * 256 * 2);
  _Float16* sW1t = (_Float16*)alloc((size_t)128 * 256 * 2);
  _Float16* tW0t = (_Float16*)alloc((size_t)256 * 256 * 2);
  _Float16* tW1t = (_Float16*)alloc((size_t)128 * 256 * 2);
  _Float16* sPW1t = (_Float16*)alloc((size_t)512 * 128 * 2);
  _Float16* sPW2t = (_Float16*)alloc((size_t)128 * 512 * 2);
  _Float16* tPW1t = (_Float16*)alloc((size_t)512 * 128 * 2);
  _Float16* tPW2t = (_Float16*)alloc((size_t)128 * 512 * 2);
  (void)ws_size; (void)out_size; (void)n_in;

  float* emb  = (float*)d_out;            // N x 128
  float* loss = (float*)d_out + (size_t)NN * 128;

  // ---- pre-transpose + f16-convert all weights ----
  wtrans_kernel<<<cdiv(256 * 256, 256), 256, 0, stream>>>(sEnc[0].W, sW0t, 256, 256);
  wtrans_kernel<<<cdiv(256 * 128, 256), 256, 0, stream>>>(sEnc[1].W, sW1t, 256, 128);
  wtrans_kernel<<<cdiv(256 * 256, 256), 256, 0, stream>>>(tEnc[0].W, tW0t, 256, 256);
  wtrans_kernel<<<cdiv(256 * 128, 256), 256, 0, stream>>>(tEnc[1].W, tW1t, 256, 128);
  wtrans_kernel<<<cdiv(128 * 512, 256), 256, 0, stream>>>(sProj.W1, sPW1t, 128, 512);
  wtrans_kernel<<<cdiv(512 * 128, 256), 256, 0, stream>>>(sProj.W2, sPW2t, 512, 128);
  wtrans_kernel<<<cdiv(128 * 512, 256), 256, 0, stream>>>(tProj.W1, tPW1t, 128, 512);
  wtrans_kernel<<<cdiv(512 * 128, 256), 256, 0, stream>>>(tProj.W2, tPW2t, 512, 128);

  auto runEncoder = [&](const EncL L[2], const _Float16* Wt0, const _Float16* Wt1,
                        const float* xin, const int* edges, float* outFinal) {
    const int* src = edges;
    const int* dst = edges + EE;
    // degree -> D^-1/2 (self loops included)
    set_kernel<<<cdiv(NN, 256), 256, 0, stream>>>(dinv, 1.0f, NN);
    deg_edge_kernel<<<cdiv(EE, 256), 256, 0, stream>>>(dst, dinv, EE);
    rsqrt_kernel<<<cdiv(NN, 256), 256, 0, stream>>>(dinv, NN);
    // layer 0: 256 -> 256
    f2h_kernel<<<cdiv(NN * 256, 256), 256, 0, stream>>>(xin, hA, NN * 256);
    wmma_gemm_kernel<<<dim3(cdiv(NN, 64), cdiv(256, 128)), 256, 0, stream>>>(
        hA, Wt0, nullptr, xw, NN, 256, 256);
    agg_self_kernel<<<cdiv(NN * 256, 256), 256, 0, stream>>>(xw, dinv, L[0].b, agg, NN, 256);
    agg_edge_kernel<<<cdiv(EE * 256, 256), 256, 0, stream>>>(xw, dinv, src, dst, agg, EE, 256);
    bn_stats_kernel<<<256 / 32, 256, 0, stream>>>(agg, mean, var, NN, 256);
    bn_prelu_kernel<<<cdiv(NN * 256, 256), 256, 0, stream>>>(agg, mean, var, L[0].gamma,
                                                             L[0].beta, L[0].a, NN, 256);
    // layer 1: 256 -> 128, output to outFinal
    f2h_kernel<<<cdiv(NN * 256, 256), 256, 0, stream>>>(agg, hA, NN * 256);
    wmma_gemm_kernel<<<dim3(cdiv(NN, 64), 1), 256, 0, stream>>>(
        hA, Wt1, nullptr, xw, NN, 256, 128);
    agg_self_kernel<<<cdiv(NN * 128, 256), 256, 0, stream>>>(xw, dinv, L[1].b, outFinal, NN, 128);
    agg_edge_kernel<<<cdiv(EE * 128, 256), 256, 0, stream>>>(xw, dinv, src, dst, outFinal, EE, 128);
    bn_stats_kernel<<<128 / 32, 256, 0, stream>>>(outFinal, mean, var, NN, 128);
    bn_prelu_kernel<<<cdiv(NN * 128, 256), 256, 0, stream>>>(outFinal, mean, var, L[1].gamma,
                                                             L[1].beta, L[1].a, NN, 128);
  };

  // three encoder passes (xw/agg buffers are free between passes)
  runEncoder(sEnc, sW0t, sW1t, x1, ei1, hencS);
  runEncoder(tEnc, tW0t, tW1t, x2, ei2, hencT);
  runEncoder(sEnc, sW0t, sW1t, x,  ei,  emb);     // emb written directly to d_out

  auto runProjector = [&](const ProjP& P, const _Float16* W1t, const _Float16* W2t,
                          const float* hin, float* outp) {
    f2h_kernel<<<cdiv(NN * 128, 256), 256, 0, stream>>>(hin, hA, NN * 128);
    wmma_gemm_kernel<<<dim3(cdiv(NN, 64), cdiv(512, 128)), 256, 0, stream>>>(
        hA, W1t, P.b1, hid, NN, 128, 512);
    bn_stats_kernel<<<512 / 32, 256, 0, stream>>>(hid, mean, var, NN, 512);
    bn_prelu_kernel<<<cdiv(NN * 512, 256), 256, 0, stream>>>(hid, mean, var, P.gamma,
                                                             P.beta, P.a, NN, 512);
    f2h_kernel<<<cdiv(NN * 512, 256), 256, 0, stream>>>(hid, hA, NN * 512);
    wmma_gemm_kernel<<<dim3(cdiv(NN, 64), 1), 256, 0, stream>>>(
        hA, W2t, P.b2, outp, NN, 512, 128);
  };
  runProjector(sProj, sPW1t, sPW2t, hencS, h1);
  runProjector(tProj, tPW1t, tPW2t, hencT, h2);

  // ---- contrastive loss: fused exp/rowsum similarity GEMMs (no N x N in memory) ----
  l2norm_half_kernel<<<cdiv(NN, 8), 256, 0, stream>>>(h1, hz1, NN, 128);
  l2norm_half_kernel<<<cdiv(NN, 8), 256, 0, stream>>>(h2, hz2, NN, 128);
  set_kernel<<<cdiv(7 * NN, 256), 256, 0, stream>>>(lvec, 0.0f, 7 * NN);

  dim3 simGrid(cdiv(NN, 64), cdiv(NN, 128));
  sim_exp_kernel<<<simGrid, 256, 0, stream>>>(hz1, hz1, rs1, nullptr, dg1, NN);  // refl1
  sim_exp_kernel<<<simGrid, 256, 0, stream>>>(hz2, hz2, rs2, nullptr, dg2, NN);  // refl2
  sim_exp_kernel<<<simGrid, 256, 0, stream>>>(hz1, hz2, rsb, csb, dgb, NN);      // betw (+colsums)

  loss_kernel<<<1, 256, 0, stream>>>(rs1, dg1, rs2, dg2, rsb, csb, dgb, loss, NN);
}
